// Graph_Processer_Physic_Predictor_43860206027001
// MI455X (gfx1250) — compile-verified, matrix operand order checked by reading, not run on hardware
//
#include <hip/hip_runtime.h>
#include <hip/hip_bf16.h>
#include <math.h>

// ---------------------------------------------------------------------------
// Types for CDNA5 WMMA (wave32): 16x16x32 f16 -> f32 accumulate.
// ---------------------------------------------------------------------------
typedef __attribute__((ext_vector_type(16))) _Float16 v16h;
typedef __attribute__((ext_vector_type(8)))  _Float16 v8h;
typedef __attribute__((ext_vector_type(8)))  float    v8f;

__device__ __forceinline__ v8f wmma_f16(v16h a, v16h b, v8f c) {
    // 8 args: (neg_a, A, neg_b, B, c_mod, C, reuse_a, reuse_b)
    return __builtin_amdgcn_wmma_f32_16x16x32_f16(false, a, false, b,
                                                  (short)0, c, false, false);
}

// K index held by fragment element e (0..15) for lane-group g (0/1),
// per CDNA5 ISA 16-bit A/B 16x32 striping:
//   e<8  -> K = g*8 + e          (VGPR 0..3)
//   e>=8 -> K = 16 + g*8 + (e-8) (VGPR 4..7)
__device__ __host__ __forceinline__ int kmap(int e, int g) {
    return (e < 8) ? (g * 8 + e) : (16 + g * 8 + (e - 8));
}

// A-fragment from a row-major f16 row in LDS: two contiguous 16B chunks.
__device__ __forceinline__ v16h load_a_row(const _Float16* row, int kbase, int g) {
    v8h lo = *(const v8h*)(row + kbase + g * 8);
    v8h hi = *(const v8h*)(row + kbase + 16 + g * 8);
    v16h a;
#pragma unroll
    for (int e = 0; e < 8; ++e) { a[e] = lo[e]; a[e + 8] = hi[e]; }
    return a;
}

// B-fragment from pre-packed weights: one contiguous 32B load per lane.
// Packed layout: ((nt*KT + kt)*32 + lane)*16 + e
__device__ __forceinline__ v16h load_b_packed(const _Float16* wp, int KT,
                                              int nt, int kt, int lane) {
    return *(const v16h*)(wp + ((size_t)(nt * KT + kt) * 32 + lane) * 16);
}

// ---------------------------------------------------------------------------
// Weight pack: row-major f32 (Ksrc x Nsrc) -> f16 WMMA B-fragment order,
// zero-padded to (KT*32 x NT*16).
// ---------------------------------------------------------------------------
__global__ void pack_b_f16(const float* __restrict__ src, _Float16* __restrict__ dst,
                           int Ksrc, int Nsrc, int KT, int NT) {
    int idx = blockIdx.x * blockDim.x + threadIdx.x;
    int total = KT * NT * 32;
    if (idx >= total) return;
    int lane = idx & 31;
    int t    = idx >> 5;
    int kt   = t % KT;
    int nt   = t / KT;
    int g    = lane >> 4;
    int n    = nt * 16 + (lane & 15);
    _Float16* out = dst + (size_t)idx * 16;
#pragma unroll
    for (int e = 0; e < 16; ++e) {
        int k = kt * 32 + kmap(e, g);
        float v = (k < Ksrc && n < Nsrc) ? src[(size_t)k * Nsrc + n] : 0.0f;
        out[e] = (_Float16)v;
    }
}

// ---------------------------------------------------------------------------
// Kernel A: fused relation MLP + masked receiver aggregation.
// One block per (batch b, receiver i): 63 relations (+1 zero pad row).
//   rel_data(64x32) --W1--> relu hidden(64x256, LDS f16) --W2--> effects(64x128)
//   agg[b,i,:] = sum_rows(mask * effects)
// 128 threads = 4 waves; wave w owns rows 16w..16w+15.
// ---------------------------------------------------------------------------
#define HSTRIDE 264   // 256 + 8 halves pad (bank-conflict avoidance)
#define ASTRIDE 40    // 32 + 8 halves pad

__global__ __launch_bounds__(128) void relation_kernel(
    const float* __restrict__ state,   // (B,64,8) raw
    const float* __restrict__ shape,   // (B,64,10) raw
    const float* __restrict__ relinfo, // (B,4032,3)
    const float* __restrict__ s_mean, const float* __restrict__ s_std,
    const float* __restrict__ sh_mean, const float* __restrict__ sh_std,
    const _Float16* __restrict__ W1p, const float* __restrict__ b1,
    const _Float16* __restrict__ W2p, const float* __restrict__ b2,
    float* __restrict__ agg)           // (B,64,128)
{
    __shared__ float    ds[64][8];
    __shared__ float    dsh[64][10];
    __shared__ _Float16 relA[64][ASTRIDE];
    __shared__ float    maskf[64];
    __shared__ _Float16 H[64][HSTRIDE];
    __shared__ float    partial[4][128];

    const int blk = blockIdx.x;
    const int b = blk >> 6;
    const int i = blk & 63;
    const int tid = threadIdx.x;

    // Normalize object state/shape for this batch into LDS.
    for (int idx = tid; idx < 64 * 8; idx += 128) {
        int n = idx >> 3, c = idx & 7;
        ds[n][c] = (state[((size_t)b * 64 + n) * 8 + c] - s_mean[c]) / s_std[c];
    }
    for (int idx = tid; idx < 64 * 10; idx += 128) {
        int n = idx / 10, c = idx % 10;
        dsh[n][c] = (shape[((size_t)b * 64 + n) * 10 + c] - sh_mean[c]) / sh_std[c];
    }
    __syncthreads();

    // Relation features: thread r (<64) builds row r of rel_data (32 cols).
    if (tid < 64) {
        int r = tid;
        if (r < 63) {
            int j = (r < i) ? r : r + 1;
            float dx = state[((size_t)b * 64 + i) * 8 + 0] - state[((size_t)b * 64 + j) * 8 + 0];
            float dy = state[((size_t)b * 64 + i) * 8 + 1] - state[((size_t)b * 64 + j) * 8 + 1];
            float dist = sqrtf(dx * dx + dy * dy);
            float m = (dist < 0.35f && dist != 0.0f) ? 1.0f : 0.0f;
            maskf[r] = m;
            float ss[8], rs[8], d[8];
#pragma unroll
            for (int c = 0; c < 8; ++c) {
                ss[c] = m * ds[j][c];
                rs[c] = m * ds[i][c];
                d[c]  = ss[c] - rs[c];
            }
            float a = rs[2], ca = cosf(a), sa = sinf(a);
            float feat[32];
            size_t rg = ((size_t)b * 4032 + (size_t)i * 63 + r) * 3;
            feat[0] = relinfo[rg + 0]; feat[1] = relinfo[rg + 1]; feat[2] = relinfo[rg + 2];
            feat[3] =  ca * d[0] + sa * d[1];     // tfed[0,0]
            feat[4] = -sa * d[0] + ca * d[1];     // tfed[1,0]
            feat[5] =  ca * d[3] + sa * d[4];     // tfed[0,1]
            feat[6] = -sa * d[3] + ca * d[4];     // tfed[1,1]
            feat[7] =  ca * ss[6] + sa * ss[7];   // tfed[0,2]
            feat[8] = -sa * ss[6] + ca * ss[7];   // tfed[1,2]
            feat[9]  = sinf(2.0f * d[2]);
            feat[10] = cosf(2.0f * d[2]);
            feat[11] = d[5];
#pragma unroll
            for (int c = 0; c < 10; ++c) {
                feat[12 + c] = m * dsh[i][c];   // receivers_shape
                feat[22 + c] = m * dsh[j][c];   // senders_shape
            }
#pragma unroll
            for (int c = 0; c < 32; ++c) relA[r][c] = (_Float16)feat[c];
        } else {
            maskf[63] = 0.0f;
#pragma unroll
            for (int c = 0; c < 32; ++c) relA[63][c] = (_Float16)0.0f;
        }
    }
    __syncthreads();

    const int wv = tid >> 5, lane = tid & 31, g = lane >> 4, ln = lane & 15;
    const _Float16* arow = &relA[wv * 16 + ln][0];

    // GEMM1: hidden(64x256) = relu(rel_data @ W1 + b1), K=32 -> 1 WMMA/tile.
#pragma unroll
    for (int nt = 0; nt < 16; ++nt) {
        v8f acc = {};
        v16h af = load_a_row(arow, 0, g);
        v16h bf = load_b_packed(W1p, 1, nt, 0, lane);
        acc = wmma_f16(af, bf, acc);
        int col = nt * 16 + ln;
        float bias = b1[col];
#pragma unroll
        for (int v = 0; v < 8; ++v) {
            int row = wv * 16 + v + 8 * g;
            H[row][col] = (_Float16)fmaxf(acc[v] + bias, 0.0f);
        }
    }
    __syncthreads();

    const _Float16* hrow = &H[wv * 16 + ln][0];

    // GEMM2: effects(64x128) = hidden @ W2 + b2; then mask + column-sum.
#pragma unroll
    for (int nt = 0; nt < 8; ++nt) {
        v8f acc = {};
#pragma unroll
        for (int kt = 0; kt < 8; ++kt) {
            v16h af = load_a_row(hrow, kt * 32, g);
            v16h bf = load_b_packed(W2p, 8, nt, kt, lane);
            acc = wmma_f16(af, bf, acc);
        }
        int col = nt * 16 + ln;
        float bias = b2[col];
        float s = 0.0f;
#pragma unroll
        for (int v = 0; v < 8; ++v) {
            int row = wv * 16 + v + 8 * g;
            s += (acc[v] + bias) * maskf[row];
        }
        s += __shfl_xor(s, 16, 32);        // combine lane-groups g=0/1
        if (g == 0) partial[wv][col] = s;
    }
    __syncthreads();

    // Reduce across the 4 waves and emit agg[b,i,:].
    {
        float s = partial[0][tid] + partial[1][tid] + partial[2][tid] + partial[3][tid];
        agg[((size_t)b * 64 + i) * 128 + tid] = s;
    }
}

// ---------------------------------------------------------------------------
// Kernel B: object MLP (141 padded to 160 -> 256 -> 3) + output compose.
// One block per batch (64 object rows); 128 threads = 4 waves.
// ---------------------------------------------------------------------------
#define XSTRIDE 168   // 160 + 8 halves pad

__global__ __launch_bounds__(128) void object_kernel(
    const float* __restrict__ state, const float* __restrict__ shape,
    const float* __restrict__ s_mean, const float* __restrict__ s_std,
    const float* __restrict__ sh_mean, const float* __restrict__ sh_std,
    const float* __restrict__ v_mean, const float* __restrict__ v_std,
    const float* __restrict__ agg,
    const _Float16* __restrict__ W3p, const float* __restrict__ b3,
    const _Float16* __restrict__ W4p, const float* __restrict__ b4,
    float* __restrict__ out)           // (B,63,6)
{
    __shared__ _Float16 X[64][XSTRIDE];
    __shared__ _Float16 H2[64][HSTRIDE];
    __shared__ float    outv[64][3];

    const int b = blockIdx.x;
    const int tid = threadIdx.x;

    // Build obj_data (13 features) per row, plus K zero pad [141..160).
    if (tid < 64) {
        int n = tid;
        float dsl[8];
#pragma unroll
        for (int c = 0; c < 8; ++c)
            dsl[c] = (state[((size_t)b * 64 + n) * 8 + c] - s_mean[c]) / s_std[c];
        float a = dsl[2], ca = cosf(a), sa = sinf(a);
        X[n][0] = (_Float16)( ca * dsl[3] + sa * dsl[4]);
        X[n][1] = (_Float16)(-sa * dsl[3] + ca * dsl[4]);
        X[n][2] = (_Float16)dsl[5];
#pragma unroll
        for (int c = 0; c < 10; ++c)
            X[n][3 + c] = (_Float16)((shape[((size_t)b * 64 + n) * 10 + c] - sh_mean[c]) / sh_std[c]);
#pragma unroll
        for (int c = 141; c < 160; ++c) X[n][c] = (_Float16)0.0f;
    }
    // agg fills cols [13..141).
    for (int idx = tid; idx < 64 * 128; idx += 128) {
        int n = idx >> 7, c = idx & 127;
        X[n][13 + c] = (_Float16)agg[((size_t)b * 64 + n) * 128 + c];
    }
    __syncthreads();

    const int wv = tid >> 5, lane = tid & 31, g = lane >> 4, ln = lane & 15;
    const _Float16* xrow = &X[wv * 16 + ln][0];

    // GEMM3: H2(64x256) = relu(X @ W3 + b3), K=160 -> 5 WMMA/tile.
#pragma unroll
    for (int nt = 0; nt < 16; ++nt) {
        v8f acc = {};
#pragma unroll
        for (int kt = 0; kt < 5; ++kt) {
            v16h af = load_a_row(xrow, kt * 32, g);
            v16h bf = load_b_packed(W3p, 5, nt, kt, lane);
            acc = wmma_f16(af, bf, acc);
        }
        int col = nt * 16 + ln;
        float bias = b3[col];
#pragma unroll
        for (int v = 0; v < 8; ++v) {
            int row = wv * 16 + v + 8 * g;
            H2[row][col] = (_Float16)fmaxf(acc[v] + bias, 0.0f);
        }
    }
    __syncthreads();

    const _Float16* hrow = &H2[wv * 16 + ln][0];

    // GEMM4: out_vel(64x3) via a single 16-wide N tile (cols 0..2 valid).
    {
        v8f acc = {};
#pragma unroll
        for (int kt = 0; kt < 8; ++kt) {
            v16h af = load_a_row(hrow, kt * 32, g);
            v16h bf = load_b_packed(W4p, 8, 0, kt, lane);
            acc = wmma_f16(af, bf, acc);
        }
        if (ln < 3) {
            float bias = b4[ln];
#pragma unroll
            for (int v = 0; v < 8; ++v) {
                int row = wv * 16 + v + 8 * g;
                outv[row][ln] = acc[v] + bias;
            }
        }
    }
    __syncthreads();

    // Final: un-rotate velocity, integrate, write x_next (rows n=1..63).
    if (tid < 64 && tid >= 1) {
        int n = tid;
        float p0 = outv[n][0] * v_std[0] + v_mean[0];
        float p1 = outv[n][1] * v_std[1] + v_mean[1];
        float p2 = outv[n][2] * v_std[2] + v_mean[2];
        float a = state[((size_t)b * 64 + n) * 8 + 2];
        float ca = cosf(a), sa = sinf(a);
        float xd0 = ca * p0 - sa * p1;   // rot_basis(-a) @ pred[:2]
        float xd1 = sa * p0 + ca * p1;
        float xd[3] = { xd0, xd1, p2 };
        float* o = out + ((size_t)b * 63 + (n - 1)) * 6;
#pragma unroll
        for (int c = 0; c < 6; ++c) {
            float base = (c < 3) ? state[((size_t)b * 64 + n) * 8 + c] : 0.0f;
            o[c] = base + xd[c % 3];
        }
    }
}

// ---------------------------------------------------------------------------
// Launch. Inputs (setup_inputs order):
//  0 objects_state (64,64,8)   1 objects_shape (64,64,10)  2 relation_info (64,4032,3)
//  3 s_mean(8)  4 s_std(8)  5 sh_mean(10)  6 sh_std(10)  7 v_mean(3)  8 v_std(3)
//  9 W1(32,256) 10 b1(256) 11 W2(256,128) 12 b2(128)
// 13 W3(141,256) 14 b3(256) 15 W4(256,3) 16 b4(3)
// Output: (64,63,6) f32.
// ---------------------------------------------------------------------------
extern "C" void kernel_launch(void* const* d_in, const int* in_sizes, int n_in,
                              void* d_out, int out_size, void* d_ws, size_t ws_size,
                              hipStream_t stream) {
    (void)in_sizes; (void)n_in; (void)out_size; (void)ws_size;

    char* ws = (char*)d_ws;
    float*    agg = (float*)ws;                              // 4096*128 f32 = 2 MB
    _Float16* W1p = (_Float16*)(ws + 4096 * 128 * sizeof(float));
    _Float16* W2p = W1p + 1 * 16 * 32 * 16;                  // 8192 halves
    _Float16* W3p = W2p + 8 * 8 * 32 * 16;                   // 32768 halves
    _Float16* W4p = W3p + 5 * 16 * 32 * 16;                  // 40960 halves
                                                             // W4p: 8*1*32*16 = 4096 halves

    // Pack weights into WMMA B-fragment order (f16).
    pack_b_f16<<<(1 * 16 * 32 + 127) / 128, 128, 0, stream>>>((const float*)d_in[9],  W1p,  32, 256, 1, 16);
    pack_b_f16<<<(8 * 8  * 32 + 127) / 128, 128, 0, stream>>>((const float*)d_in[11], W2p, 256, 128, 8,  8);
    pack_b_f16<<<(5 * 16 * 32 + 127) / 128, 128, 0, stream>>>((const float*)d_in[13], W3p, 141, 256, 5, 16);
    pack_b_f16<<<(8 * 1  * 32 + 127) / 128, 128, 0, stream>>>((const float*)d_in[15], W4p, 256,   3, 8,  1);

    relation_kernel<<<64 * 64, 128, 0, stream>>>(
        (const float*)d_in[0], (const float*)d_in[1], (const float*)d_in[2],
        (const float*)d_in[3], (const float*)d_in[4],
        (const float*)d_in[5], (const float*)d_in[6],
        W1p, (const float*)d_in[10], W2p, (const float*)d_in[12], agg);

    object_kernel<<<64, 128, 0, stream>>>(
        (const float*)d_in[0], (const float*)d_in[1],
        (const float*)d_in[3], (const float*)d_in[4],
        (const float*)d_in[5], (const float*)d_in[6],
        (const float*)d_in[7], (const float*)d_in[8],
        agg, W3p, (const float*)d_in[14], W4p, (const float*)d_in[16],
        (float*)d_out);
}
